// GraphNeuralNetwork_69947837382982
// MI455X (gfx1250) — matrix operand order repaired and checked by reading, not compiled, not run on hardware
//
#include <hip/hip_runtime.h>
#include <hip/hip_bf16.h>

// ---------------- problem constants ----------------
#define NN     20000
#define EE     60000
#define BB     512
#define F_IN   7
#define ED     3
#define HH     4
#define CC     256
#define HC     1024
#define D_MODEL 1024
#define NTOT   (EE + NN)           // edges + self loops
#define KDIM   1024

typedef __attribute__((ext_vector_type(16))) _Float16 v16h;
typedef __attribute__((ext_vector_type(8)))  _Float16 v8h;
typedef __attribute__((ext_vector_type(8)))  float    v8f;

// ---------- helpers ----------
__device__ __forceinline__ unsigned enc_f32(float f) {
  unsigned b = __float_as_uint(f);
  return (b & 0x80000000u) ? ~b : (b | 0x80000000u);
}
__device__ __forceinline__ float dec_f32(unsigned k) {
  unsigned b = (k & 0x80000000u) ? (k ^ 0x80000000u) : ~k;
  return __uint_as_float(b);
}

// ================= weight packing: f32 [3][1024][1024] -> f16 B-fragment order =================
// layout: wpack[l][ntile(64)][ktile(32)][lane(32)][16 halves]
// lane<16: N=n0+lane, K=k0+0..7,16..23 ; lane>=16: N=n0+lane-16, K=k0+8..15,24..31
__global__ __launch_bounds__(256) void pack_w_k(const float* __restrict__ w_rest,
                                                _Float16* __restrict__ wpack) {
  int idx = blockIdx.x * 256 + threadIdx.x;          // ((l*64+nt)*32+kt)*32+lane
  if (idx >= 3 * 64 * 32 * 32) return;
  int lane = idx & 31;
  int kt   = (idx >> 5) & 31;
  int nt   = (idx >> 10) & 63;
  int l    = idx >> 16;
  const float* W = w_rest + (size_t)l * KDIM * HC;
  int n = nt * 16 + (lane & 15);
  int kbase = kt * 32 + (lane >> 4) * 8;
  _Float16* dst = wpack + (size_t)idx * 16;
#pragma unroll
  for (int i = 0; i < 16; ++i) dst[i] = (_Float16)W[(size_t)(kbase + i + (i >= 8 ? 8 : 0)) * HC + n];
}

// ================= main WMMA GEMM: C[M,1024] = A_f16[M,1024] x Wpack =================
__device__ __forceinline__ v16h load_a_frag(const _Float16* p) {
  v8h lo = *(const v8h*)p;
  v8h hi = *(const v8h*)(p + 16);
  v16h a;
#pragma unroll
  for (int i = 0; i < 8; ++i) { a[i] = lo[i]; a[i + 8] = hi[i]; }
  return a;
}

__global__ __launch_bounds__(256) void gemm_wmma_k(const _Float16* __restrict__ A,
                                                   const _Float16* __restrict__ Bp,
                                                   float* __restrict__ C) {
  const int wave = threadIdx.x >> 5;          // 0..7
  const int lane = threadIdx.x & 31;
  const int m0 = blockIdx.x * 16;             // grid.x = 1250
  const int n0 = blockIdx.y * 512 + wave * 64;// grid.y = 2
  const int row16 = lane & 15, kg = lane >> 4;

  v8f acc0 = {}, acc1 = {}, acc2 = {}, acc3 = {};
  const _Float16* arow = A + (size_t)(m0 + row16) * KDIM + kg * 8;
  const int nt0 = n0 >> 4;

  for (int kt = 0; kt < KDIM / 32; ++kt) {
    v16h a = load_a_frag(arow + kt * 32);
    const _Float16* bbase = Bp + ((size_t)nt0 * (KDIM / 32) + kt) * 32 * 16 + (size_t)lane * 16;
    const size_t nstride = (size_t)(KDIM / 32) * 32 * 16;   // halves per ntile
    v16h b0 = *(const v16h*)(bbase);
    v16h b1 = *(const v16h*)(bbase + nstride);
    v16h b2 = *(const v16h*)(bbase + 2 * nstride);
    v16h b3 = *(const v16h*)(bbase + 3 * nstride);
    acc0 = __builtin_amdgcn_wmma_f32_16x16x32_f16(false, a, false, b0, (short)0, acc0, false, false);
    acc1 = __builtin_amdgcn_wmma_f32_16x16x32_f16(false, a, false, b1, (short)0, acc1, false, false);
    acc2 = __builtin_amdgcn_wmma_f32_16x16x32_f16(false, a, false, b2, (short)0, acc2, false, false);
    acc3 = __builtin_amdgcn_wmma_f32_16x16x32_f16(false, a, false, b3, (short)0, acc3, false, false);
  }
  // C/D layout: vgpr j, lanes0-15: M=j,N=lane ; lanes16-31: M=8+j,N=lane-16
  const int col = lane & 15;
  const int rbase = m0 + (lane >> 4) * 8;
  float* c0 = C + (size_t)rbase * HC + n0 + col;
#pragma unroll
  for (int j = 0; j < 8; ++j) {
    float* cr = c0 + (size_t)j * HC;
    cr[0]  = acc0[j];
    cr[16] = acc1[j];
    cr[32] = acc2[j];
    cr[48] = acc3[j];
  }
}

// ================= layer-0 GEMM (K=7, trivial) =================
__global__ __launch_bounds__(256) void gemm_l0_k(const float* __restrict__ x,
                                                 const float* __restrict__ w0,
                                                 float* __restrict__ out) {
  int idx = blockIdx.x * 256 + threadIdx.x;   // n*1024+j
  if (idx >= NN * HC) return;
  int n = idx >> 10, j = idx & 1023;
  float s = 0.f;
#pragma unroll
  for (int d = 0; d < F_IN; ++d) s += x[n * F_IN + d] * w0[d * HC + j];
  out[idx] = s;
}

// ================= edge-attr mean (single block, deterministic) =================
__global__ __launch_bounds__(256) void ea_mean_k(const float* __restrict__ ea, float* __restrict__ eam) {
  __shared__ float r0[256], r1[256], r2[256];
  int t = threadIdx.x;
  float s0 = 0, s1 = 0, s2 = 0;
  for (int e = t; e < EE; e += 256) { s0 += ea[e*3]; s1 += ea[e*3+1]; s2 += ea[e*3+2]; }
  r0[t] = s0; r1[t] = s1; r2[t] = s2; __syncthreads();
  for (int off = 128; off >= 1; off >>= 1) {
    if (t < off) { r0[t] += r0[t+off]; r1[t] += r1[t+off]; r2[t] += r2[t+off]; }
    __syncthreads();
  }
  if (t == 0) { eam[0] = r0[0] / EE; eam[1] = r1[0] / EE; eam[2] = r2[0] / EE; }
}

// ================= M[l][d][h] = sum_c w_edge[l][d][h*256+c] * att_edge[l][h][c] =================
__global__ __launch_bounds__(256) void medge_k(const float* __restrict__ w_edge,
                                               const float* __restrict__ att_edge,
                                               float* __restrict__ M) {
  int bid = blockIdx.x;                 // l*12 + d*4 + h, 48 blocks
  int l = bid / 12, r = bid % 12, d = r / 4, h = r % 4;
  int t = threadIdx.x;
  float p = w_edge[(size_t)(l*3 + d) * HC + h * CC + t] * att_edge[(size_t)(l*4 + h) * CC + t];
  __shared__ float red[256];
  red[t] = p; __syncthreads();
  for (int off = 128; off >= 1; off >>= 1) { if (t < off) red[t] += red[t+off]; __syncthreads(); }
  if (t == 0) M[bid] = red[0];
}

// ================= alpha_e for all 4 layers: [4][NTOT][4] =================
__global__ __launch_bounds__(256) void alpha_e_k(const float* __restrict__ ea,
                                                 const float* __restrict__ eam,
                                                 const float* __restrict__ M,
                                                 float* __restrict__ alphaE) {
  int idx = blockIdx.x * 256 + threadIdx.x;
  if (idx >= 4 * NTOT) return;
  int l = idx / NTOT, e = idx - l * NTOT;
  float a0, a1, a2;
  if (e < EE) { a0 = ea[e*3]; a1 = ea[e*3+1]; a2 = ea[e*3+2]; }
  else        { a0 = eam[0];  a1 = eam[1];    a2 = eam[2]; }
  const float* Ml = M + l * 12;
  float* o = alphaE + (size_t)idx * 4;
#pragma unroll
  for (int h = 0; h < 4; ++h) o[h] = a0 * Ml[h] + a1 * Ml[4+h] + a2 * Ml[8+h];
}

// ================= per-node attention dots: alpha_src/dst [N][H] =================
__global__ __launch_bounds__(256) void node_alpha_k(const float* __restrict__ hlin,
                                                    const float* __restrict__ att_src,
                                                    const float* __restrict__ att_dst,
                                                    float* __restrict__ asrc,
                                                    float* __restrict__ adst, int layer) {
  int n = blockIdx.x, t = threadIdx.x;
  const float* hr = hlin + (size_t)n * HC + t * 4;
  const float* as = att_src + (size_t)layer * HC + t * 4;
  const float* ad = att_dst + (size_t)layer * HC + t * 4;
  float ps = 0, pd = 0;
#pragma unroll
  for (int k = 0; k < 4; ++k) { float v = hr[k]; ps += v * as[k]; pd += v * ad[k]; }
  __shared__ float rs[256], rd[256];
  rs[t] = ps; rd[t] = pd; __syncthreads();
  for (int off = 32; off >= 1; off >>= 1) {
    if ((t & 63) < off) { rs[t] += rs[t+off]; rd[t] += rd[t+off]; }
    __syncthreads();
  }
  if ((t & 63) == 0) { int h = t >> 6; asrc[n*4 + h] = rs[t]; adst[n*4 + h] = rd[t]; }
}

// ================= attention pass1: leaky_relu + segment max =================
__global__ __launch_bounds__(256) void att_pass1_k(const int* __restrict__ ei,
                                                   const float* __restrict__ asrc,
                                                   const float* __restrict__ adst,
                                                   const float* __restrict__ alphaE,
                                                   float* __restrict__ apre,
                                                   unsigned* __restrict__ amax, int layer) {
  int e = blockIdx.x * 256 + threadIdx.x;
  if (e >= NTOT) return;
  int src, dst;
  if (e < EE) { src = ei[e]; dst = ei[EE + e]; } else { src = dst = e - EE; }
  const float* ae = alphaE + ((size_t)layer * NTOT + e) * 4;
#pragma unroll
  for (int h = 0; h < 4; ++h) {
    float a = asrc[src*4 + h] + adst[dst*4 + h] + ae[h];
    a = a > 0.f ? a : 0.2f * a;
    apre[e*4 + h] = a;
    atomicMax(&amax[dst*4 + h], enc_f32(a));
  }
}

// ================= attention pass2: exp + segment sum =================
__global__ __launch_bounds__(256) void att_pass2_k(const int* __restrict__ ei,
                                                   float* __restrict__ apre,
                                                   const unsigned* __restrict__ amax,
                                                   float* __restrict__ denom) {
  int e = blockIdx.x * 256 + threadIdx.x;
  if (e >= NTOT) return;
  int dst = (e < EE) ? ei[EE + e] : e - EE;
#pragma unroll
  for (int h = 0; h < 4; ++h) {
    float m = dec_f32(amax[dst*4 + h]);
    float w = __expf(apre[e*4 + h] - m);
    apre[e*4 + h] = w;
    atomicAdd(&denom[dst*4 + h], w);
  }
}

// ================= message scatter: out[dst] += w * hlin[src] =================
__global__ __launch_bounds__(256) void scatter_msg_k(const int* __restrict__ ei,
                                                     const float* __restrict__ hlin,
                                                     const float* __restrict__ apre,
                                                     const float* __restrict__ denom,
                                                     float* __restrict__ out) {
  int e = blockIdx.x;
  int src, dst;
  if (e < EE) { src = ei[e]; dst = ei[EE + e]; } else { src = dst = e - EE; }
  float wh[4];
#pragma unroll
  for (int h = 0; h < 4; ++h) wh[h] = apre[e*4 + h] / (denom[dst*4 + h] + 1e-16f);
  const float* hs = hlin + (size_t)src * HC;
  float* od = out + (size_t)dst * HC;
  for (int j = threadIdx.x; j < HC; j += 256)
    atomicAdd(&od[j], wh[j >> 8] * hs[j]);
}

// ================= post (layers 0-2): bias + BN + ReLU -> f16 =================
__global__ __launch_bounds__(256) void post_bn_relu_k(const float* __restrict__ agg,
                                                      const float* __restrict__ bias012,
                                                      const float* __restrict__ gamma,
                                                      const float* __restrict__ beta,
                                                      const float* __restrict__ mean,
                                                      const float* __restrict__ var,
                                                      _Float16* __restrict__ hf16, int layer) {
  int idx = blockIdx.x * 256 + threadIdx.x;
  if (idx >= NN * HC) return;
  int j = idx & 1023;
  int o = layer * HC + j;
  float v = agg[idx] + bias012[o];
  float sc = gamma[o] * rsqrtf(var[o] + 1e-5f);
  v = (v - mean[o]) * sc + beta[o];
  v = v > 0.f ? v : 0.f;
  hf16[idx] = (_Float16)v;
}

// ================= post (layer 3): head mean + bias3 -> [N,256] f32 =================
__global__ __launch_bounds__(256) void post_l3_k(const float* __restrict__ agg,
                                                 const float* __restrict__ bias3,
                                                 float* __restrict__ hf) {
  int idx = blockIdx.x * 256 + threadIdx.x;   // n*256+c
  if (idx >= NN * CC) return;
  int n = idx >> 8, c = idx & 255;
  const float* a = agg + (size_t)n * HC + c;
  hf[idx] = 0.25f * (a[0] + a[256] + a[512] + a[768]) + bias3[c];
}

// ================= gate score + segment max over batch =================
__global__ __launch_bounds__(256) void gate_score_k(const float* __restrict__ hf,
                                                    const float* __restrict__ gw,
                                                    const float* __restrict__ gb,
                                                    const int* __restrict__ batch,
                                                    float* __restrict__ gate,
                                                    unsigned* __restrict__ gmax) {
  int n = blockIdx.x, t = threadIdx.x;
  float p = hf[(size_t)n * CC + t] * gw[t];
  __shared__ float red[256];
  red[t] = p; __syncthreads();
  for (int off = 128; off >= 1; off >>= 1) { if (t < off) red[t] += red[t+off]; __syncthreads(); }
  if (t == 0) {
    float g = red[0] + gb[0];
    gate[n] = g;
    atomicMax(&gmax[batch[n]], enc_f32(g));
  }
}

__global__ __launch_bounds__(256) void gate_pass2_k(float* __restrict__ gate,
                                                    const unsigned* __restrict__ gmax,
                                                    float* __restrict__ gsum,
                                                    const int* __restrict__ batch) {
  int n = blockIdx.x * 256 + threadIdx.x;
  if (n >= NN) return;
  int b = batch[n];
  float g = __expf(gate[n] - dec_f32(gmax[b]));
  gate[n] = g;
  atomicAdd(&gsum[b], g);
}

__global__ __launch_bounds__(256) void graph_accum_k(const float* __restrict__ hf,
                                                     const float* __restrict__ gate,
                                                     const float* __restrict__ gsum,
                                                     const int* __restrict__ batch,
                                                     float* __restrict__ graph) {
  int n = blockIdx.x, t = threadIdx.x;
  int b = batch[n];
  float w = gate[n] / (gsum[b] + 1e-16f);
  atomicAdd(&graph[b * CC + t], w * hf[(size_t)n * CC + t]);
}

// ================= final projection: [B,256] x [256,1024] + bias =================
__global__ __launch_bounds__(256) void proj_k(const float* __restrict__ graph,
                                              const float* __restrict__ pw,
                                              const float* __restrict__ pb,
                                              float* __restrict__ out) {
  int idx = blockIdx.x * 256 + threadIdx.x;   // b*1024+j
  if (idx >= BB * D_MODEL) return;
  int b = idx >> 10, j = idx & 1023;
  float s = pb[j];
  for (int c = 0; c < CC; ++c) s += graph[b * CC + c] * pw[c * D_MODEL + j];
  out[idx] = s;
}

// =======================================================================
extern "C" void kernel_launch(void* const* d_in, const int* in_sizes, int n_in,
                              void* d_out, int out_size, void* d_ws, size_t ws_size,
                              hipStream_t stream) {
  const float* x        = (const float*)d_in[0];
  const int*   ei       = (const int*)  d_in[1];
  const float* ea       = (const float*)d_in[2];
  const int*   batch    = (const int*)  d_in[3];
  const float* w0       = (const float*)d_in[4];
  const float* w_rest   = (const float*)d_in[5];
  const float* w_edge   = (const float*)d_in[6];
  const float* att_src  = (const float*)d_in[7];
  const float* att_dst  = (const float*)d_in[8];
  const float* att_edge = (const float*)d_in[9];
  const float* bias012  = (const float*)d_in[10];
  const float* bias3    = (const float*)d_in[11];
  const float* bn_gamma = (const float*)d_in[12];
  const float* bn_beta  = (const float*)d_in[13];
  const float* bn_mean  = (const float*)d_in[14];
  const float* bn_var   = (const float*)d_in[15];
  const float* gate_w   = (const float*)d_in[16];
  const float* gate_b   = (const float*)d_in[17];
  const float* proj_w   = (const float*)d_in[18];
  const float* proj_b   = (const float*)d_in[19];
  float* out = (float*)d_out;

  // ---- carve workspace ----
  char* ws = (char*)d_ws;
  size_t off = 0;
  auto carve = [&](size_t bytes) { void* p = ws + off; off = (off + bytes + 255) & ~(size_t)255; return p; };
  float*     bufA   = (float*)    carve((size_t)NN * HC * 4);      // GEMM out (h_lin)
  float*     bufB   = (float*)    carve((size_t)NN * HC * 4);      // aggregation out
  _Float16*  hf16   = (_Float16*) carve((size_t)NN * HC * 2);      // next-layer GEMM input
  _Float16*  wpack  = (_Float16*) carve((size_t)3 * 64 * 32 * 32 * 16 * 2);
  float*     alphaE = (float*)    carve((size_t)4 * NTOT * 4 * 4);
  float*     asrc   = (float*)    carve((size_t)NN * 4 * 4);
  float*     adst   = (float*)    carve((size_t)NN * 4 * 4);
  unsigned*  amax   = (unsigned*) carve((size_t)NN * 4 * 4);
  float*     denom  = (float*)    carve((size_t)NN * 4 * 4);
  float*     apre   = (float*)    carve((size_t)NTOT * 4 * 4);
  float*     eam    = (float*)    carve(256);
  float*     Medge  = (float*)    carve(256);
  float*     hfin   = (float*)    carve((size_t)NN * CC * 4);
  float*     gate   = (float*)    carve((size_t)NN * 4);
  unsigned*  gmax   = (unsigned*) carve((size_t)BB * 4);
  float*     gsum   = (float*)    carve((size_t)BB * 4);
  float*     graph  = (float*)    carve((size_t)BB * CC * 4);
  (void)ws_size; (void)in_sizes; (void)n_in; (void)out_size;

  // ---- one-time (per call) precompute ----
  pack_w_k<<<(3*64*32*32 + 255)/256, 256, 0, stream>>>(w_rest, wpack);
  ea_mean_k<<<1, 256, 0, stream>>>(ea, eam);
  medge_k<<<48, 256, 0, stream>>>(w_edge, att_edge, Medge);
  alpha_e_k<<<(4*NTOT + 255)/256, 256, 0, stream>>>(ea, eam, Medge, alphaE);

  // ---- 4 GAT layers ----
  for (int l = 0; l < 4; ++l) {
    if (l == 0) {
      gemm_l0_k<<<(NN*HC)/256, 256, 0, stream>>>(x, w0, bufA);
    } else {
      gemm_wmma_k<<<dim3(NN/16, 2), 256, 0, stream>>>(
          hf16, wpack + (size_t)(l - 1) * 64 * 32 * 32 * 16, bufA);
    }
    node_alpha_k<<<NN, 256, 0, stream>>>(bufA, att_src, att_dst, asrc, adst, l);
    hipMemsetAsync(amax,  0, (size_t)NN * 4 * 4, stream);
    hipMemsetAsync(denom, 0, (size_t)NN * 4 * 4, stream);
    hipMemsetAsync(bufB,  0, (size_t)NN * HC * 4, stream);
    att_pass1_k<<<(NTOT + 255)/256, 256, 0, stream>>>(ei, asrc, adst, alphaE, apre, amax, l);
    att_pass2_k<<<(NTOT + 255)/256, 256, 0, stream>>>(ei, apre, amax, denom);
    scatter_msg_k<<<NTOT, 256, 0, stream>>>(ei, bufA, apre, denom, bufB);
    if (l < 3) {
      post_bn_relu_k<<<(NN*HC)/256, 256, 0, stream>>>(bufB, bias012, bn_gamma, bn_beta,
                                                      bn_mean, bn_var, hf16, l);
    } else {
      post_l3_k<<<(NN*CC)/256, 256, 0, stream>>>(bufB, bias3, hfin);
    }
  }

  // ---- attentional aggregation + projection ----
  hipMemsetAsync(gmax,  0, (size_t)BB * 4, stream);
  hipMemsetAsync(gsum,  0, (size_t)BB * 4, stream);
  hipMemsetAsync(graph, 0, (size_t)BB * CC * 4, stream);
  gate_score_k<<<NN, 256, 0, stream>>>(hfin, gate_w, gate_b, batch, gate, gmax);
  gate_pass2_k<<<(NN + 255)/256, 256, 0, stream>>>(gate, gmax, gsum, batch);
  graph_accum_k<<<NN, 256, 0, stream>>>(hfin, gate, gsum, batch, graph);
  proj_k<<<(BB*D_MODEL)/256, 256, 0, stream>>>(graph, proj_w, proj_b, out);
}
